// Attend_60181081752257
// MI455X (gfx1250) — matrix operand order repaired and checked by reading, not compiled, not run on hardware
//
#include <hip/hip_runtime.h>
#include <hip/hip_bf16.h>

typedef _Float16 v16h __attribute__((ext_vector_type(16)));
typedef _Float16 v8h  __attribute__((ext_vector_type(8)));
typedef float    v8f  __attribute__((ext_vector_type(8)));

// Raw v_exp_f32 (computes 2^x on AMD hardware).
__device__ __forceinline__ float exp2_hw(float x) {
    return __builtin_amdgcn_exp2f(x);
}

// Concatenate two 16-byte half-vectors into one WMMA 16-half fragment.
__device__ __forceinline__ v16h load2x8(const _Float16* p0, const _Float16* p1) {
    v8h a = *(const v8h*)p0;
    v8h b = *(const v8h*)p1;
    return __builtin_shufflevector(a, b, 0,1,2,3,4,5,6,7,8,9,10,11,12,13,14,15);
}

// ---------------------------------------------------------------------------
// Prepass: f32 -> f16, producing both row-major (dh: [B][R][C]) and
// transposed (dt: [B][C][R]) copies via an LDS tile so all global traffic is
// coalesced. R=rows (1024), C=cols (128).
// ---------------------------------------------------------------------------
__global__ __launch_bounds__(1024) void conv_kernel(
    const float* __restrict__ src, _Float16* __restrict__ dh,
    _Float16* __restrict__ dt, int R, int C)
{
    __shared__ _Float16 tile[32][33];
    const int b  = blockIdx.z;
    const int r0 = blockIdx.x * 32;
    const int c0 = blockIdx.y * 32;
    const int tx = threadIdx.x, ty = threadIdx.y;

    const size_t base = (size_t)b * R * C;
    float v = src[base + (size_t)(r0 + ty) * C + (c0 + tx)];
    _Float16 h = (_Float16)v;
    dh[base + (size_t)(r0 + ty) * C + (c0 + tx)] = h;
    tile[ty][tx] = h;
    __syncthreads();
    dt[(size_t)b * C * R + (size_t)(c0 + ty) * R + (r0 + tx)] = tile[tx][ty];
}

// ---------------------------------------------------------------------------
// Flash attention (no scaling): Out[b,i,:] = softmax_j(Q[b,i]·K[b,j]) · V[b,j]
//   Qh: [B][M][128] f16 row-major   (Q A-fragments, d-contiguous)
//   Kh: [B][N][128] f16 row-major   (K staged to LDS [key][d])
//   Vt: [B][128][N] f16 d-major     (V staged to LDS [d][key])
//   Out:[B][M][128] f32
// 256-thread blocks: 8 waves share one LDS-staged 64-key K/V chunk
// (8x less L2 traffic than per-wave streaming); each wave owns 16 query rows.
// Softmax in log2 domain (Q pre-scaled by log2 e -> raw v_exp_f32); row sums
// via ones-column WMMA (lane-replicated in C-layout); only the row max needs
// a 16-lane shuffle tree. P is re-laid out C->A format through per-wave LDS.
// ---------------------------------------------------------------------------
__global__ __launch_bounds__(256) void flash_kernel(
    const _Float16* __restrict__ Qh, const _Float16* __restrict__ Kh,
    const _Float16* __restrict__ Vt, float* __restrict__ Out,
    int M, int N)
{
    constexpr int D     = 128;
    constexpr int PADK  = 136;   // halves per K-chunk row (272B: bank-staggered, 16B-aligned)
    constexpr int PADV  = 72;    // halves per V-chunk row (144B)
    constexpr int PITCH = 72;    // halves per P row (144B)

    __shared__ _Float16 Klds[64 * PADK];        // [key 0..63][d 0..127]
    __shared__ _Float16 Vlds[128 * PADV];       // [d 0..127][key 0..63]
    __shared__ _Float16 pbuf_all[8 * 16 * PITCH];

    const int tid    = threadIdx.x;
    const int wave   = tid >> 5;
    const int lane   = tid & 31;
    const int lane16 = lane & 15;
    const int half16 = lane >> 4;

    const int b    = blockIdx.x >> 3;
    const int qblk = ((blockIdx.x & 7) << 3) | wave;    // 0..63
    const int q0   = qblk * 16;

    _Float16* pbuf = pbuf_all + wave * 16 * PITCH;

    const _Float16* Qb = Qh + (size_t)b * M * D;
    const _Float16* Kb = Kh + (size_t)b * N * D;
    const _Float16* Vb = Vt + (size_t)b * D * N;

    // ---- Q A-fragments (16 rows x 32 d) x 4 chunks, pre-scaled by log2(e) ----
    v16h qf[4];
    {
        v16h l2e;
        #pragma unroll
        for (int i = 0; i < 16; ++i) l2e[i] = (_Float16)1.44269504088896340736f;
        const _Float16* qrow = Qb + (size_t)(q0 + lane16) * D + half16 * 8;
        #pragma unroll
        for (int dc = 0; dc < 4; ++dc)
            qf[dc] = load2x8(qrow + dc * 32, qrow + dc * 32 + 16) * l2e;
    }

    v16h ones;
    #pragma unroll
    for (int i = 0; i < 16; ++i) ones[i] = (_Float16)1.0f;

    // ---- accumulators ----
    v8f o[8];
    #pragma unroll
    for (int t = 0; t < 8; ++t) o[t] = (v8f){};
    float mrow[8], lrow[8];
    #pragma unroll
    for (int r = 0; r < 8; ++r) { mrow[r] = -INFINITY; lrow[r] = 0.0f; }

    // Chunk-staging indices (each thread moves 4 v8h of K and 4 of V = 128B).
    // K chunk: 64 keys x 128 halves = 1024 v8h; V chunk: 128 d x 64 halves.
    int kkey[4], kcol[4], vrow[4], vcol[4];
    #pragma unroll
    for (int i = 0; i < 4; ++i) {
        int fk = tid + 256 * i;
        kkey[i] = fk >> 4;  kcol[i] = (fk & 15) * 8;
        vrow[i] = fk >> 3;  vcol[i] = (fk & 7) * 8;
    }

    // ---- prefetch chunk 0 into registers ----
    v8h kreg[4], vreg[4];
    #pragma unroll
    for (int i = 0; i < 4; ++i) {
        kreg[i] = *(const v8h*)(Kb + (size_t)kkey[i] * D + kcol[i]);
        vreg[i] = *(const v8h*)(Vb + (size_t)vrow[i] * N + vcol[i]);
    }

    for (int kk = 0; kk < N; kk += 64) {
        __syncthreads();                         // all waves done with prev chunk
        #pragma unroll
        for (int i = 0; i < 4; ++i) {
            *(v8h*)(Klds + kkey[i] * PADK + kcol[i]) = kreg[i];
            *(v8h*)(Vlds + vrow[i] * PADV + vcol[i]) = vreg[i];
        }
        __syncthreads();                         // staged chunk visible

        // prefetch next chunk while computing on this one
        if (kk + 64 < N) {
            #pragma unroll
            for (int i = 0; i < 4; ++i) {
                kreg[i] = *(const v8h*)(Kb + (size_t)(kk + 64 + kkey[i]) * D + kcol[i]);
                vreg[i] = *(const v8h*)(Vb + (size_t)vrow[i] * N + kk + 64 + vcol[i]);
            }
        }

        // ---- S = (log2e * Q) · K^T for 64 keys: 4 C-tiles ----
        v8f s[4];
        #pragma unroll
        for (int ti = 0; ti < 4; ++ti) s[ti] = (v8f){};
        #pragma unroll
        for (int dc = 0; dc < 4; ++dc) {
            #pragma unroll
            for (int ti = 0; ti < 4; ++ti) {
                const _Float16* kp =
                    Klds + (16 * ti + lane16) * PADK + dc * 32 + half16 * 8;
                v16h kf = load2x8(kp, kp + 16);
                s[ti] = __builtin_amdgcn_wmma_f32_16x16x32_f16(false, qf[dc], false, kf,
                                                               (short)0, s[ti], false, false);
            }
        }

        // ---- row max (only remaining shuffle tree) ----
        float scale[8], newm[8];
        #pragma unroll
        for (int r = 0; r < 8; ++r) {
            float t = fmaxf(fmaxf(s[0][r], s[1][r]), fmaxf(s[2][r], s[3][r]));
            t = fmaxf(t, __shfl_xor(t, 1, 32));
            t = fmaxf(t, __shfl_xor(t, 2, 32));
            t = fmaxf(t, __shfl_xor(t, 4, 32));
            t = fmaxf(t, __shfl_xor(t, 8, 32));
            float nm = fmaxf(mrow[r], t);
            scale[r] = exp2_hw(mrow[r] - nm);     // 2^(m_old - m_new)
            mrow[r] = nm;
            newm[r] = nm;
        }

        // ---- P = 2^(S - m), written straight to LDS as f16 in C-layout ----
        #pragma unroll
        for (int ti = 0; ti < 4; ++ti)
            #pragma unroll
            for (int r = 0; r < 8; ++r)
                pbuf[(r + half16 * 8) * PITCH + ti * 16 + lane16] =
                    (_Float16)exp2_hw(s[ti][r] - newm[r]);

        asm volatile("s_wait_dscnt 0" ::: "memory");   // wave-local LDS RAW fence

        // ---- reload P as two A-fragments (16 rows x 32 keys each) ----
        const _Float16* prow = pbuf + lane16 * PITCH + half16 * 8;
        v16h pf0 = load2x8(prow,      prow + 16);      // keys kk+ 0..31
        v16h pf1 = load2x8(prow + 32, prow + 48);      // keys kk+32..63

        // ---- row sums via ones-column WMMA: result replicated across lanes ----
        v8f lt = (v8f){};
        lt = __builtin_amdgcn_wmma_f32_16x16x32_f16(false, pf0, false, ones,
                                                    (short)0, lt, false, false);
        lt = __builtin_amdgcn_wmma_f32_16x16x32_f16(false, pf1, false, ones,
                                                    (short)0, lt, false, false);

        // ---- rescale running state ----
        #pragma unroll
        for (int r = 0; r < 8; ++r) lrow[r] = lrow[r] * scale[r] + lt[r];
        #pragma unroll
        for (int t = 0; t < 8; ++t)
            #pragma unroll
            for (int r = 0; r < 8; ++r) o[t][r] *= scale[r];

        // ---- O += P · V (8 d-tiles x 2 key-halves) from LDS ----
        #pragma unroll
        for (int t = 0; t < 8; ++t) {
            const _Float16* vp = Vlds + (16 * t + lane16) * PADV + half16 * 8;
            v16h vf0 = load2x8(vp,      vp + 16);      // keys kk+ 0..31
            v16h vf1 = load2x8(vp + 32, vp + 48);      // keys kk+32..63
            o[t] = __builtin_amdgcn_wmma_f32_16x16x32_f16(false, pf0, false, vf0,
                                                          (short)0, o[t], false, false);
            o[t] = __builtin_amdgcn_wmma_f32_16x16x32_f16(false, pf1, false, vf1,
                                                          (short)0, o[t], false, false);
        }
    }

    // ---- normalize and store (row = q0 + r + 8*half16, col = 16t + lane16) ----
    #pragma unroll
    for (int r = 0; r < 8; ++r) lrow[r] = 1.0f / lrow[r];
    float* ob = Out + (size_t)b * M * D;
    #pragma unroll
    for (int t = 0; t < 8; ++t)
        #pragma unroll
        for (int r = 0; r < 8; ++r)
            ob[(size_t)(q0 + r + half16 * 8) * D + t * 16 + lane16] = o[t][r] * lrow[r];
}

// ---------------------------------------------------------------------------
extern "C" void kernel_launch(void* const* d_in, const int* in_sizes, int n_in,
                              void* d_out, int out_size, void* d_ws, size_t ws_size,
                              hipStream_t stream) {
    (void)in_sizes; (void)n_in; (void)out_size; (void)ws_size;
    constexpr int BATCH = 32, M = 1024, N = 1024, D = 128;
    const size_t elems = (size_t)BATCH * M * D;   // 4,194,304

    const float* A  = (const float*)d_in[0];
    const float* Bm = (const float*)d_in[1];
    float* out = (float*)d_out;

    // workspace: Ah | Bh | At | Bt  (f16 each, 8 MB each = 32 MB total)
    _Float16* Ah = (_Float16*)d_ws;
    _Float16* Bh = Ah + elems;
    _Float16* At = Bh + elems;
    _Float16* Bt = At + elems;

    dim3 cblk(32, 32, 1);
    conv_kernel<<<dim3(M / 32, D / 32, BATCH), cblk, 0, stream>>>(A,  Ah, At, M, D);
    conv_kernel<<<dim3(N / 32, D / 32, BATCH), cblk, 0, stream>>>(Bm, Bh, Bt, N, D);

    // beta : Q=A, K=V=B, softmax over n
    flash_kernel<<<dim3(BATCH * 8), dim3(256), 0, stream>>>(Ah, Bh, Bt, out, M, N);
    // alpha: Q=B, K=V=A, softmax over m (e transposed)
    flash_kernel<<<dim3(BATCH * 8), dim3(256), 0, stream>>>(Bh, Ah, At, out + elems, N, M);
}
